// GATClassic_1013612282532
// MI455X (gfx1250) — matrix operand order repaired and checked by reading, not compiled, not run on hardware
//
#include <hip/hip_runtime.h>
#include <hip/hip_bf16.h>

typedef __attribute__((ext_vector_type(16))) _Float16 v16h;
typedef __attribute__((ext_vector_type(8)))  _Float16 v8h;
typedef __attribute__((ext_vector_type(8)))  float    v8f;

#define GAT_D 128
#define WT_RS 136   // padded row stride (halfs) for transposed W in LDS: 272B, 16B-aligned
#define NEG_SLOPE 0.2f
#define EPS_DEN 1e-16f

// ---------------------------------------------------------------------------
// Kernel 1: H = X @ W via v_wmma_f32_16x16x32_f16.
// Block = 256 threads = 8 waves; computes a 64-row x 128-col slab of H.
// LDS: W TRANSPOSED as f16 (Wt[n][k], 128 x WT_RS ~ 34KB) + X tile (16KB).
// Both A and B fragments are contiguous 16B chunks -> ds_load_b128.
// Wave w: rows (w>>1)*16..+16, cols (w&1)*64..+64 (4 16x16 tiles, 4 v8f accs).
// ---------------------------------------------------------------------------
__global__ __launch_bounds__(256) void gat_gemm_wmma(
    const float* __restrict__ X, const float* __restrict__ W,
    float* __restrict__ H, int N)
{
    __shared__ _Float16 Wt[GAT_D * WT_RS];   // transposed: Wt[n*WT_RS + k]
    __shared__ _Float16 Xl[64 * GAT_D];      // row-major X tile

    const int tid  = threadIdx.x;
    const int row0 = blockIdx.x * 64;

    // Stage W -> LDS transposed (coalesced global float4 reads, strided LDS
    // u16 stores; one-time cost per block).
    for (int i = tid * 4; i < GAT_D * GAT_D; i += 256 * 4) {
        float4 w4 = *(const float4*)(W + i);
        int r = i >> 7;          // k index
        int c = i & (GAT_D - 1); // n index
        Wt[(c + 0) * WT_RS + r] = (_Float16)w4.x;
        Wt[(c + 1) * WT_RS + r] = (_Float16)w4.y;
        Wt[(c + 2) * WT_RS + r] = (_Float16)w4.z;
        Wt[(c + 3) * WT_RS + r] = (_Float16)w4.w;
    }
    // Stage X tile -> LDS; zero-pad rows beyond N.
    for (int i = tid * 4; i < 64 * GAT_D; i += 256 * 4) {
        int r = row0 + (i >> 7);
        int c = i & (GAT_D - 1);
        if (r < N) {
            float4 x4 = *(const float4*)(X + (size_t)r * GAT_D + c);
            Xl[i + 0] = (_Float16)x4.x; Xl[i + 1] = (_Float16)x4.y;
            Xl[i + 2] = (_Float16)x4.z; Xl[i + 3] = (_Float16)x4.w;
        } else {
            Xl[i + 0] = (_Float16)0.f; Xl[i + 1] = (_Float16)0.f;
            Xl[i + 2] = (_Float16)0.f; Xl[i + 3] = (_Float16)0.f;
        }
    }
    __syncthreads();

    const int wave  = tid >> 5;
    const int lane  = tid & 31;
    const int half  = lane >> 4;    // 0/1: K-half selector for A/B frags
    const int ln    = lane & 15;    // M for A frag, N for B/C/D frags
    const int mbase = (wave >> 1) * 16;   // row tile within slab
    const int nbase = (wave & 1) * 64;    // col group base

    v8f acc[4] = {};

#pragma unroll
    for (int kc = 0; kc < GAT_D; kc += 32) {
        // A fragment (16x32 f16): lane (half, m=ln), element e:
        //   K = kc + 8*half + e (+8 if e>=8)  -> two contiguous 16B chunks.
        const _Float16* ap = &Xl[(mbase + ln) * GAT_D + kc + 8 * half];
        v8h alo = *(const v8h*)(ap);
        v8h ahi = *(const v8h*)(ap + 16);
        v16h a = __builtin_shufflevector(alo, ahi,
                 0,1,2,3,4,5,6,7,8,9,10,11,12,13,14,15);
#pragma unroll
        for (int nt = 0; nt < 4; ++nt) {
            // B fragment (32x16 f16): lane (half, n=ln), element e:
            //   K = kc + 8*half + e (+8 if e>=8)  -> two contiguous 16B chunks in Wt row n.
            const _Float16* bp =
                &Wt[(nbase + nt * 16 + ln) * WT_RS + kc + 8 * half];
            v8h blo = *(const v8h*)(bp);
            v8h bhi = *(const v8h*)(bp + 16);
            v16h b = __builtin_shufflevector(blo, bhi,
                     0,1,2,3,4,5,6,7,8,9,10,11,12,13,14,15);
            acc[nt] = __builtin_amdgcn_wmma_f32_16x16x32_f16(
                false, a, false, b, (short)0, acc[nt], false, false);
        }
    }

    // D layout: VGPR r, lane (half, n=ln) -> row = r + 8*half, col = n.
#pragma unroll
    for (int nt = 0; nt < 4; ++nt) {
#pragma unroll
        for (int r = 0; r < 8; ++r) {
            int row = row0 + mbase + r + 8 * half;
            if (row < N)
                H[(size_t)row * GAT_D + nbase + nt * 16 + ln] = acc[nt][r];
        }
    }
}

// ---------------------------------------------------------------------------
// Kernel 2: s[n] = H[n]·a_src, t[n] = H[n]·a_dst.  One wave32 per node,
// 4 floats per lane, shuffle reduction.
// ---------------------------------------------------------------------------
__global__ __launch_bounds__(256) void gat_st(
    const float* __restrict__ H, const float* __restrict__ asrc,
    const float* __restrict__ adst, float* __restrict__ s,
    float* __restrict__ t, int N)
{
    int gw   = (blockIdx.x * blockDim.x + threadIdx.x) >> 5;
    int lane = threadIdx.x & 31;
    if (gw >= N) return;
    const float* hrow = H + (size_t)gw * GAT_D;
    float4 h4 = *(const float4*)(hrow + lane * 4);
    float4 a4 = *(const float4*)(asrc + lane * 4);
    float4 d4 = *(const float4*)(adst + lane * 4);
    float ps = h4.x * a4.x + h4.y * a4.y + h4.z * a4.z + h4.w * a4.w;
    float pt = h4.x * d4.x + h4.y * d4.y + h4.z * d4.z + h4.w * d4.w;
#pragma unroll
    for (int off = 16; off > 0; off >>= 1) {
        ps += __shfl_down(ps, off, 32);
        pt += __shfl_down(pt, off, 32);
    }
    if (lane == 0) { s[gw] = ps; t[gw] = pt; }
}

// ---------------------------------------------------------------------------
// Kernel 3: zero out accumulator, reset segment-max encodings and denoms.
// ---------------------------------------------------------------------------
__global__ void gat_init(float* __restrict__ out, unsigned* __restrict__ menc,
                         float* __restrict__ denom, int N)
{
    int i = blockIdx.x * blockDim.x + threadIdx.x;
    if (i < N) { menc[i] = 0u; denom[i] = 0.f; }
    if (i < N * GAT_D) out[i] = 0.f;
}

__device__ __forceinline__ unsigned enc_f32(float v) {
    unsigned b = __float_as_uint(v);
    return (b & 0x80000000u) ? ~b : (b | 0x80000000u);
}
__device__ __forceinline__ float dec_f32(unsigned u) {
    unsigned b = (u & 0x80000000u) ? (u & 0x7FFFFFFFu) : ~u;
    return __uint_as_float(b);
}

// ---------------------------------------------------------------------------
// Kernel 4: per-edge logits e = leaky_relu(s[src]+t[dst]); segment max via
// monotonic-uint atomicMax. Edges i>=E are self-loops (src=dst=i-E).
// ---------------------------------------------------------------------------
__global__ void gat_edge_logits(const int* __restrict__ ei, int E, int N,
                                const float* __restrict__ s,
                                const float* __restrict__ t,
                                float* __restrict__ eval,
                                unsigned* __restrict__ menc)
{
    int i = blockIdx.x * blockDim.x + threadIdx.x;
    if (i >= E + N) return;
    int src = (i < E) ? ei[i]     : (i - E);
    int dst = (i < E) ? ei[E + i] : (i - E);
    float v = s[src] + t[dst];
    v = (v > 0.f) ? v : NEG_SLOPE * v;
    eval[i] = v;
    atomicMax(menc + dst, enc_f32(v));
}

// ---------------------------------------------------------------------------
// Kernel 5: ex = exp(e - m[dst]) in place; denom[dst] += ex.
// ---------------------------------------------------------------------------
__global__ void gat_edge_exp(const int* __restrict__ ei, int E, int N,
                             const unsigned* __restrict__ menc,
                             float* __restrict__ eval,
                             float* __restrict__ denom)
{
    int i = blockIdx.x * blockDim.x + threadIdx.x;
    if (i >= E + N) return;
    int dst = (i < E) ? ei[E + i] : (i - E);
    float ex = __expf(eval[i] - dec_f32(menc[dst]));
    eval[i] = ex;
    atomicAdd(denom + dst, ex);
}

// ---------------------------------------------------------------------------
// Kernel 6: weighted scatter. One wave per edge: alpha = ex/(denom+eps);
// out[dst][:] += alpha * h[src][:].  float4 per lane, 4 f32 atomics per lane.
// out/denom/h are L2-resident (192 MB L2), so atomics resolve on-chip.
// ---------------------------------------------------------------------------
__global__ __launch_bounds__(256) void gat_scatter(
    const int* __restrict__ ei, int E, int N,
    const float* __restrict__ H, const float* __restrict__ eval,
    const float* __restrict__ denom, float* __restrict__ out)
{
    int gw   = (blockIdx.x * blockDim.x + threadIdx.x) >> 5;
    int lane = threadIdx.x & 31;
    if (gw >= E + N) return;
    int src = (gw < E) ? ei[gw]     : (gw - E);
    int dst = (gw < E) ? ei[E + gw] : (gw - E);
    float alpha = eval[gw] / (denom[dst] + EPS_DEN);
    const float* hrow = H + (size_t)src * GAT_D;
    float*       orow = out + (size_t)dst * GAT_D;
    float4 h4 = *(const float4*)(hrow + lane * 4);
    atomicAdd(orow + lane * 4 + 0, h4.x * alpha);
    atomicAdd(orow + lane * 4 + 1, h4.y * alpha);
    atomicAdd(orow + lane * 4 + 2, h4.z * alpha);
    atomicAdd(orow + lane * 4 + 3, h4.w * alpha);
}

// ---------------------------------------------------------------------------
// Kernel 7: out = relu(out + bias)
// ---------------------------------------------------------------------------
__global__ void gat_bias_relu(float* __restrict__ out,
                              const float* __restrict__ bias, int N)
{
    int i = blockIdx.x * blockDim.x + threadIdx.x;
    if (i >= N * GAT_D) return;
    float v = out[i] + bias[i & (GAT_D - 1)];
    out[i] = (v > 0.f) ? v : 0.f;
}

// ---------------------------------------------------------------------------
// Host launcher: two independent GAT layers; workspace reused sequentially.
// ---------------------------------------------------------------------------
extern "C" void kernel_launch(void* const* d_in, const int* in_sizes, int n_in,
                              void* d_out, int out_size, void* d_ws, size_t ws_size,
                              hipStream_t stream)
{
    const int N = in_sizes[0] / GAT_D;   // 50000
    const int E = in_sizes[1] / 2;       // 600000
    const int Etot = E + N;

    // Workspace layout (reused for both graphs):
    char* ws = (char*)d_ws;
    float*    h     = (float*)ws;                      ws += (size_t)N * GAT_D * sizeof(float);
    float*    s     = (float*)ws;                      ws += (size_t)N * sizeof(float);
    float*    t     = (float*)ws;                      ws += (size_t)N * sizeof(float);
    unsigned* menc  = (unsigned*)ws;                   ws += (size_t)N * sizeof(unsigned);
    float*    denom = (float*)ws;                      ws += (size_t)N * sizeof(float);
    float*    eval  = (float*)ws;                      ws += (size_t)Etot * sizeof(float);

    for (int g = 0; g < 2; ++g) {
        const float* x    = (const float*)d_in[g == 0 ? 0 : 3];
        const int*   ei   = (const int*)  d_in[g == 0 ? 1 : 4];
        const float* W    = (const float*)d_in[g == 0 ? 6 : 10];
        const float* asrc = (const float*)d_in[g == 0 ? 7 : 11];
        const float* adst = (const float*)d_in[g == 0 ? 8 : 12];
        const float* bias = (const float*)d_in[g == 0 ? 9 : 13];
        float* out = (float*)d_out + (size_t)g * N * GAT_D;

        gat_init<<<(N * GAT_D + 255) / 256, 256, 0, stream>>>(out, menc, denom, N);
        gat_gemm_wmma<<<(N + 63) / 64, 256, 0, stream>>>(x, W, h, N);
        gat_st<<<(N * 32 + 255) / 256, 256, 0, stream>>>(h, asrc, adst, s, t, N);
        gat_edge_logits<<<(Etot + 255) / 256, 256, 0, stream>>>(ei, E, N, s, t, eval, menc);
        gat_edge_exp<<<(Etot + 255) / 256, 256, 0, stream>>>(ei, E, N, menc, eval, denom);
        gat_scatter<<<(Etot * 32 + 255) / 256, 256, 0, stream>>>(ei, E, N, h, eval, denom, out);
        gat_bias_relu<<<(N * GAT_D + 255) / 256, 256, 0, stream>>>(out, bias, N);
    }
}
